// SoftPoolFeat_89275190215483
// MI455X (gfx1250) — matrix operand order repaired and checked by reading, not compiled, not run on hardware
//
#include <hip/hip_runtime.h>
#include <hip/hip_bf16.h>
#include <cstdint>

// ---------------------------------------------------------------------------
// SoftPool PointNet encoder for MI455X (gfx1250, wave32, WMMA + TDM).
//  - All pointwise-conv GEMMs run on V_WMMA_F32_16X16X32_BF16 (fp32 accum).
//  - W tiles staged into LDS by the Tensor Data Mover (TENSOR_LOAD_TO_LDS,
//    TENSORcnt sync), with D# padding reproducing the bank-conflict-free
//    64B+16B LDS row layout. Weights pre-converted to bf16 once.
//  - X tiles staged with vectorized loads + software transpose (TDM cannot
//    transpose; 2-byte-line descriptors would serialize the DMA).
//  - Non-temporal stores for the 268 MB write-once output stream.
//  - STN global max fused into the 1024-ch GEMM via uint-bit atomic max.
//  - 512 descending argsorts (8192 keys) as LDS bitonic sort on packed u64.
// ---------------------------------------------------------------------------

#if defined(__has_builtin)
# if __has_builtin(__builtin_amdgcn_tensor_load_to_lds) && \
     __has_builtin(__builtin_amdgcn_s_wait_tensorcnt)
#  define USE_TDM 1
# endif
#endif
#ifndef USE_TDM
# define USE_TDM 0
#endif

namespace {

constexpr int B_       = 8;
constexpr int NPTS     = 8192;
constexpr int REGIONS_ = 64;
constexpr int SP       = 2048;
constexpr float EPS_   = 1e-5f;

typedef __bf16 bf16_t;
typedef __attribute__((ext_vector_type(4)))  __bf16 bf16x4;
typedef __attribute__((ext_vector_type(8)))  __bf16 bf16x8;
typedef __attribute__((ext_vector_type(16))) __bf16 bf16x16;
typedef __attribute__((ext_vector_type(8)))  float  f32x8;

#if USE_TDM
// 2D tile load Global->LDS via Tensor Data Mover.  data_size = 2 bytes.
// LDS layout produced: tile_d1 rows of tile_d0 elements, each 64B row
// followed by 16B padding (pad_interval code 3 = 64B, pad_amount code 3 =
// 4 DWORDs) -> matches the Wt[MTILE][KC+8] LDS layout exactly.
__device__ __forceinline__ void tdm_load_tile_2d_b16(
    uint32_t lds_byte_addr, const void* gtile,
    uint32_t tensor_d0, uint32_t tensor_d1, uint32_t d0_stride_elems,
    uint32_t tile_d0, uint32_t tile_d1)
{
  typedef unsigned int u32x4 __attribute__((ext_vector_type(4)));
  typedef int          i32x4 __attribute__((ext_vector_type(4)));
  typedef int          i32x8 __attribute__((ext_vector_type(8)));
  uint64_t ga = (uint64_t)(uintptr_t)gtile;
  u32x4 g0;
  g0[0] = 1u;                                              // count=1, user D#
  g0[1] = lds_byte_addr;                                   // lds_addr[31:0]
  g0[2] = (uint32_t)ga;                                    // global_addr[31:0]
  g0[3] = (uint32_t)((ga >> 32) & 0x01FFFFFFu) | (2u << 30); // addr[56:32]|type=2
  i32x8 g1;
  g1[0] = (int)((1u << 16)       // data_size = 2 bytes
              | (1u << 20)       // pad_enable
              | (3u << 22)       // pad_interval: 64 bytes
              | (3u << 25));     // pad_amount: 4 DWORDs (16B)
  g1[1] = (int)((tensor_d0 & 0xFFFFu) << 16);              // tensor_dim0[15:0]
  g1[2] = (int)(((tensor_d0 >> 16) & 0xFFFFu) | ((tensor_d1 & 0xFFFFu) << 16));
  g1[3] = (int)(((tensor_d1 >> 16) & 0xFFFFu) | ((tile_d0 & 0xFFFFu) << 16));
  g1[4] = (int)(tile_d1 & 0xFFFFu);                        // tile_dim1; dim2=0
  g1[5] = (int)d0_stride_elems;                            // dim0_stride[31:0]
  g1[6] = 0;                                               // stride hi / dim1_stride
  g1[7] = 0;
  i32x4 z4 = {0, 0, 0, 0};
#if __clang_major__ >= 23
  i32x8 z8 = {0, 0, 0, 0, 0, 0, 0, 0};
  __builtin_amdgcn_tensor_load_to_lds(g0, g1, z4, z4, z8, 0);
#else
  __builtin_amdgcn_tensor_load_to_lds(g0, g1, z4, z4, 0);
#endif
}
#endif // USE_TDM

// ------------------------- fp32 -> bf16 weight pre-conversion ---------------
__global__ __launch_bounds__(256) void cvt_w_kernel(
    const float* __restrict__ s, bf16_t* __restrict__ d, int n4)
{
  int i = blockIdx.x * 256 + threadIdx.x;
  if (i >= n4) return;
  float4 v = ((const float4*)s)[i];
  bf16x4 o;
  o[0] = (bf16_t)v.x; o[1] = (bf16_t)v.y;
  o[2] = (bf16_t)v.z; o[3] = (bf16_t)v.w;
  ((bf16x4*)d)[i] = o;
}

// ------------------------- conv1 (K=3) + BN + ReLU -> bf16 ------------------
template<bool TRANSFORM>
__global__ __launch_bounds__(256) void conv1_bn_relu_kernel(
    const float* __restrict__ x, const float* __restrict__ trans,
    const float* __restrict__ w, const float* __restrict__ cb,
    const float* __restrict__ bg, const float* __restrict__ bb,
    const float* __restrict__ bm, const float* __restrict__ bv,
    bf16_t* __restrict__ h1)
{
  int t = blockIdx.x * 256 + threadIdx.x;       // (b, n)
  int b = t / NPTS, n = t % NPTS;
  const float* xb = x + (size_t)b * 3 * NPTS + n;
  float p0 = xb[0], p1 = xb[NPTS], p2 = xb[2 * NPTS];
  float q0 = p0, q1 = p1, q2 = p2;
  if (TRANSFORM) {
    const float* T = trans + b * 9;             // trans[b][c][d] = T[3c+d]
    q0 = p0 * T[0] + p1 * T[3] + p2 * T[6];
    q1 = p0 * T[1] + p1 * T[4] + p2 * T[7];
    q2 = p0 * T[2] + p1 * T[5] + p2 * T[8];
  }
  bf16_t* out = h1 + (size_t)b * 64 * NPTS + n;
  #pragma unroll 8
  for (int o = 0; o < 64; ++o) {
    float v = q0 * w[o * 3] + q1 * w[o * 3 + 1] + q2 * w[o * 3 + 2] + cb[o];
    float sc = bg[o] * rsqrtf(bv[o] + EPS_);
    v = fmaxf((v - bm[o]) * sc + bb[o], 0.f);
    out[(size_t)o * NPTS] = (bf16_t)v;
  }
}

// ------------------------- WMMA pointwise conv ------------------------------
// C(M,N) = W(M,K) * X(K,N) per batch, K staged through LDS in 32-deep chunks.
// MODE 0: BN+ReLU, store bf16 (B,M,N)
// MODE 1: BN+ReLU, max-reduce over N into gmax[B*M] (uint bits of f32 >= 0)
// MODE 2: BN only,  store f32 transposed (B,N,M)
constexpr int KC    = 32;
constexpr int NTILE = 128;
constexpr int LDP   = 8;     // row padding (bf16) -> 80B rows, 16B aligned

template<int MTILE, int KTOT, int MODE>
__global__ __launch_bounds__(256) void wmma_pconv_kernel(
    const bf16_t* __restrict__ W,               // bf16 weights, row-major (M,K)
    const float* __restrict__ cb,
    const float* __restrict__ bg, const float* __restrict__ bb,
    const float* __restrict__ bm, const float* __restrict__ bv,
    const bf16_t* __restrict__ X, void* __restrict__ outp,
    unsigned* __restrict__ gmax, int Mtot)
{
  __shared__ bf16_t Wt[MTILE][KC + LDP];
  __shared__ bf16_t Xt[NTILE][KC + LDP];            // transposed: [n][k]
  __shared__ unsigned redmax[MODE == 1 ? MTILE : 1];

  const int b    = blockIdx.x / (NPTS / NTILE);
  const int nb   = (blockIdx.x % (NPTS / NTILE)) * NTILE;
  const int mb   = blockIdx.y * MTILE;
  const int tid  = threadIdx.x;
  const int wave = tid >> 5, lane = tid & 31;
  const int half = lane >> 4, lm = lane & 15;

  constexpr int MSLOTS  = MTILE / 16;               // wave M-slots
  constexpr int NGROUPS = 8 / MSLOTS;
  constexpr int NT      = (NTILE / 16) / NGROUPS;   // N-subtiles per wave
  const int mslot = wave % MSLOTS;
  const int ngrp  = wave / MSLOTS;

  f32x8 acc[NT] = {};
  const bf16_t* Xb = X + (size_t)b * KTOT * NPTS;

  for (int kc = 0; kc < KTOT; kc += KC) {
#if USE_TDM
    // W tile DMA: wave 0 issues TENSOR_LOAD_TO_LDS; overlaps with the
    // cooperative X staging below; drained via TENSORcnt before the barrier.
    if (wave == 0) {
      tdm_load_tile_2d_b16((uint32_t)(uintptr_t)&Wt[0][0],
                           W + (size_t)mb * KTOT + kc,
                           (uint32_t)KTOT, (uint32_t)Mtot, (uint32_t)KTOT,
                           (uint32_t)KC, (uint32_t)MTILE);
    }
#else
    for (int i = tid; i < MTILE * KC / 8; i += 256) {
      int r = (i * 8) / KC, c = (i * 8) % KC;
      *(bf16x8*)&Wt[r][c] = *(const bf16x8*)&W[(size_t)(mb + r) * KTOT + (kc + c)];
    }
#endif
    // stage X tile transposed: 16B bf16x8 global loads, 8x b16 LDS deposit
    for (int i = tid; i < KC * NTILE / 8; i += 256) {
      int k = (i * 8) / NTILE, n0 = (i * 8) % NTILE;
      bf16x8 xv = *(const bf16x8*)&Xb[(size_t)(kc + k) * NPTS + (nb + n0)];
      #pragma unroll
      for (int e = 0; e < 8; ++e) Xt[n0 + e][k] = xv[e];
    }
#if USE_TDM
    if (wave == 0) __builtin_amdgcn_s_wait_tensorcnt(0);
#endif
    __syncthreads();

    // A fragment: lane = M row (lm); K halves split across lane halves:
    //   half h: K in [8h, 8h+8) and [16+8h, 16+8h+8)  -> two 16B LDS reads
    const int arow = mslot * 16 + lm;
    bf16x8 alo = *(const bf16x8*)&Wt[arow][8 * half];
    bf16x8 ahi = *(const bf16x8*)&Wt[arow][16 + 8 * half];
    bf16x16 afrag = __builtin_shufflevector(alo, ahi,
        0,1,2,3,4,5,6,7,8,9,10,11,12,13,14,15);

    #pragma unroll
    for (int t = 0; t < NT; ++t) {
      // B fragment: lane = N col (lm); K in [8h, 8h+8) and [16+8h, 16+8h+8) -> two 16B LDS reads
      const int ncol = (ngrp * NT + t) * 16 + lm;
      bf16x8 blo = *(const bf16x8*)&Xt[ncol][8 * half];
      bf16x8 bhi = *(const bf16x8*)&Xt[ncol][16 + 8 * half];
      bf16x16 bfrag = __builtin_shufflevector(blo, bhi,
          0,1,2,3,4,5,6,7,8,9,10,11,12,13,14,15);
      acc[t] = __builtin_amdgcn_wmma_f32_16x16x32_bf16(
          false, afrag, false, bfrag, (short)0, acc[t], false, false);
    }
    __syncthreads();
  }

  if constexpr (MODE == 1) {
    for (int i = tid; i < MTILE; i += 256) redmax[i] = 0u;
    __syncthreads();
  }

  // C/D layout: VGPR j -> row M = 8*half + j (within 16-tile), col N = lm
  float sj[8], oj[8];
  #pragma unroll
  for (int j = 0; j < 8; ++j) {
    int gr = mb + mslot * 16 + 8 * half + j;
    float sc = bg[gr] * rsqrtf(bv[gr] + EPS_);
    sj[j] = sc;
    oj[j] = (cb[gr] - bm[gr]) * sc + bb[gr];
  }
  #pragma unroll
  for (int t = 0; t < NT; ++t) {
    int gc = nb + (ngrp * NT + t) * 16 + lm;
    #pragma unroll
    for (int j = 0; j < 8; ++j) {
      int gr = mb + mslot * 16 + 8 * half + j;
      float y = acc[t][j] * sj[j] + oj[j];
      if constexpr (MODE == 0) {
        y = fmaxf(y, 0.f);
        ((bf16_t*)outp)[((size_t)b * Mtot + gr) * NPTS + gc] = (bf16_t)y;
      } else if constexpr (MODE == 1) {
        y = fmaxf(y, 0.f);
        atomicMax(&redmax[gr - mb], __float_as_uint(y));   // ds_max_u32
      } else {
        ((float*)outp)[((size_t)b * NPTS + gc) * Mtot + gr] = y;
      }
    }
  }
  if constexpr (MODE == 1) {
    __syncthreads();
    for (int i = tid; i < MTILE; i += 256)
      atomicMax(&gmax[(size_t)b * Mtot + (mb + i)], redmax[i]);
  }
}

// ------------------------- STN FC head (tiny, fp32 VALU) --------------------
template<bool BNRELU>
__global__ void fc_kernel(const float* __restrict__ in, const float* __restrict__ w,
                          const float* __restrict__ bias,
                          const float* bg, const float* bb,
                          const float* bm, const float* bv,
                          float* __restrict__ out, int O, int K, int addIden)
{
  int g = blockIdx.x * blockDim.x + threadIdx.x;
  if (g >= B_ * O) return;
  int b = g / O, o = g % O;
  const float* xr = in + (size_t)b * K;
  const float* wr = w + (size_t)o * K;
  float s = 0.f;
  for (int k = 0; k < K; k += 4) {
    s = fmaf(xr[k + 0], wr[k + 0], s);
    s = fmaf(xr[k + 1], wr[k + 1], s);
    s = fmaf(xr[k + 2], wr[k + 2], s);
    s = fmaf(xr[k + 3], wr[k + 3], s);
  }
  s += bias[o];
  if constexpr (BNRELU) {
    float sc = bg[o] * rsqrtf(bv[o] + EPS_);
    s = fmaxf((s - bm[o]) * sc + bb[o], 0.f);
  } else {
    if (addIden && (o % 4) == 0) s += 1.f;     // + eye(3) at flat 0,4,8
  }
  out[g] = s;
}

// ------------------------- SoftPool: bitonic argsort ------------------------
// Keys: (~orderkey(v) << 32) | index, so an ascending u64 sort == descending
// value with ascending-index tie-break (matches stable argsort of -feats).
__global__ __launch_bounds__(1024) void softpool_sort_kernel(
    const float* __restrict__ feats_t,        // (B, N, 64)
    unsigned* __restrict__ idxbuf,            // (B, 64, SP)
    float* __restrict__ spidx)                // (B, 3, 64, SP)
{
  __shared__ unsigned long long keys[NPTS];   // 64 KB of 320 KB WGP LDS
  const int b = blockIdx.x / REGIONS_;
  const int r = blockIdx.x % REGIONS_;
  const int tid = threadIdx.x;

  for (int n = tid; n < NPTS; n += 1024) {
    float v = feats_t[((size_t)b * NPTS + n) * REGIONS_ + r];
    unsigned u = __float_as_uint(v);
    u = (u & 0x80000000u) ? ~u : (u | 0x80000000u);  // monotone ascending map
    keys[n] = ((unsigned long long)(~u) << 32) | (unsigned)n;
  }
  __syncthreads();

  for (int ksz = 2; ksz <= NPTS; ksz <<= 1) {
    for (int j = ksz >> 1; j > 0; j >>= 1) {
      for (int i = tid; i < NPTS / 2; i += 1024) {
        int ix = (i << 1) - (i & (j - 1));
        int iy = ix + j;
        bool asc = ((ix & ksz) == 0);
        unsigned long long a = keys[ix], c = keys[iy];
        if ((a > c) == asc) { keys[ix] = c; keys[iy] = a; }
      }
      __syncthreads();
    }
  }

  const size_t cstr = (size_t)REGIONS_ * SP;
  for (int k = tid; k < SP; k += 1024) {
    unsigned n = (unsigned)keys[k];
    idxbuf[((size_t)b * REGIONS_ + r) * SP + k] = n;   // re-read by gather: RT
    float fn = (float)n;
    size_t base = ((size_t)b * 3 * REGIONS_ + r) * SP + k;
    __builtin_nontemporal_store(fn, &spidx[base]);     // write-once: NT
    __builtin_nontemporal_store(fn, &spidx[base + cstr]);
    __builtin_nontemporal_store(fn, &spidx[base + 2 * cstr]);
  }
}

// ------------------------- SoftPool: gather (HBM-bound) ---------------------
__global__ __launch_bounds__(256) void softpool_gather_kernel(
    const float* __restrict__ feats_t, const unsigned* __restrict__ idxbuf,
    float* __restrict__ out)                  // (B, 64, 64, SP)
{
  size_t t = (size_t)blockIdx.x * 256 + threadIdx.x;   // (b, r, k)
  int k = (int)(t % SP);
  int r = (int)((t / SP) % REGIONS_);
  int b = (int)(t / ((size_t)SP * REGIONS_));
  unsigned n = idxbuf[t];
  const float4* src = (const float4*)(feats_t + ((size_t)b * NPTS + n) * REGIONS_);
  const size_t cs = (size_t)REGIONS_ * SP;
  float* ob = out + (size_t)b * 64 * cs + (size_t)r * SP + k;
  #pragma unroll
  for (int cq = 0; cq < 16; ++cq) {
    float4 v = src[cq];                       // contiguous 256B channel row (L2)
    __builtin_nontemporal_store(v.x, &ob[(size_t)(4 * cq + 0) * cs]);
    __builtin_nontemporal_store(v.y, &ob[(size_t)(4 * cq + 1) * cs]);
    __builtin_nontemporal_store(v.z, &ob[(size_t)(4 * cq + 2) * cs]);
    __builtin_nontemporal_store(v.w, &ob[(size_t)(4 * cq + 3) * cs]);
  }
}

} // namespace

// ---------------------------------------------------------------------------
// d_in layout: pytree flatten of setup_inputs() (dict keys sorted, tuples in
// order): 0:x; params: 1-4 bn1(g,b,m,v), 5-8 bn2, 9-12 bn3, 13 c1b, 14 c1w,
// 15 c2b, 16 c2w, 17 c3b, 18 c3w; stn: 19-22 bn1, 23-26 bn2, 27-30 bn3,
// 31-34 bn4, 35-38 bn5, 39 c1b, 40 c1w, 41 c2b, 42 c2w, 43 c3b, 44 c3w,
// 45 f1b, 46 f1w, 47 f2b, 48 f2w, 49 f3b, 50 f3w.
// ---------------------------------------------------------------------------
extern "C" void kernel_launch(void* const* d_in, const int* in_sizes, int n_in,
                              void* d_out, int out_size, void* d_ws, size_t ws_size,
                              hipStream_t stream)
{
  auto F = [&](int i) { return (const float*)d_in[i]; };
  const float* x = F(0);
  // main params
  const float *bn1g=F(1), *bn1b=F(2), *bn1m=F(3), *bn1v=F(4);
  const float *bn2g=F(5), *bn2b=F(6), *bn2m=F(7), *bn2v=F(8);
  const float *bn3g=F(9), *bn3b=F(10), *bn3m=F(11), *bn3v=F(12);
  const float *c1b=F(13), *c1w=F(14), *c2b=F(15), *c2w=F(16), *c3b=F(17), *c3w=F(18);
  // stn params
  const float *sb1g=F(19), *sb1b=F(20), *sb1m=F(21), *sb1v=F(22);
  const float *sb2g=F(23), *sb2b=F(24), *sb2m=F(25), *sb2v=F(26);
  const float *sb3g=F(27), *sb3b=F(28), *sb3m=F(29), *sb3v=F(30);
  const float *sb4g=F(31), *sb4b=F(32), *sb4m=F(33), *sb4v=F(34);
  const float *sb5g=F(35), *sb5b=F(36), *sb5m=F(37), *sb5v=F(38);
  const float *sc1b=F(39), *sc1w=F(40), *sc2b=F(41), *sc2w=F(42);
  const float *sc3b=F(43), *sc3w=F(44);
  const float *f1b=F(45), *f1w=F(46), *f2b=F(47), *f2w=F(48), *f3b=F(49), *f3w=F(50);

  // d_out sections: out (B,64,64,SP), sp_idx (B,3,64,SP), trans (B,3,3)
  float* out_main  = (float*)d_out;
  float* out_spidx = out_main + (size_t)B_ * 64 * REGIONS_ * SP;
  float* out_trans = out_spidx + (size_t)B_ * 3 * REGIONS_ * SP;

  // workspace carve-up (~44.4 MB)
  char* ws = (char*)d_ws;
  unsigned* g_max   = (unsigned*)(ws + 0);                 // B*1024 u32 (f32 bits)
  float*    f1      = (float*)(ws + 32768);                // B*512
  float*    f2      = (float*)(ws + 49152);                // B*256
  bf16_t*   h1      = (bf16_t*)(ws + 65536);               // B*64*N bf16 (8 MB)
  bf16_t*   h2      = (bf16_t*)(ws + 8454144);             // B*128*N bf16 (16 MB)
  float*    feats_t = (float*)(ws + 25231360);             // B*N*64 f32 (16 MB)
  unsigned* idxbuf  = (unsigned*)(ws + 42008576);          // B*64*SP u32 (4 MB)
  bf16_t*   wbf_sc2 = (bf16_t*)(ws + 46202880);            // 128*64 bf16
  bf16_t*   wbf_sc3 = (bf16_t*)(ws + 46219264);            // 1024*128 bf16
  bf16_t*   wbf_c2  = (bf16_t*)(ws + 46481408);            // 128*64 bf16
  bf16_t*   wbf_c3  = (bf16_t*)(ws + 46497792);            // 64*128 bf16

  hipMemsetAsync(g_max, 0, (size_t)B_ * 1024 * sizeof(unsigned), stream);

  // pre-convert all GEMM weights to bf16 (enables TDM staging, halves W reads)
  cvt_w_kernel<<<8,   256, 0, stream>>>(sc2w, wbf_sc2, 128 * 64 / 4);
  cvt_w_kernel<<<128, 256, 0, stream>>>(sc3w, wbf_sc3, 1024 * 128 / 4);
  cvt_w_kernel<<<8,   256, 0, stream>>>(c2w,  wbf_c2,  128 * 64 / 4);
  cvt_w_kernel<<<8,   256, 0, stream>>>(c3w,  wbf_c3,  64 * 128 / 4);

  // ---- STN ----
  conv1_bn_relu_kernel<false><<<256, 256, 0, stream>>>(
      x, nullptr, sc1w, sc1b, sb1g, sb1b, sb1m, sb1v, h1);
  wmma_pconv_kernel<128, 64, 0><<<dim3(512, 1), 256, 0, stream>>>(
      wbf_sc2, sc2b, sb2g, sb2b, sb2m, sb2v, h1, h2, nullptr, 128);
  wmma_pconv_kernel<128, 128, 1><<<dim3(512, 8), 256, 0, stream>>>(
      wbf_sc3, sc3b, sb3g, sb3b, sb3m, sb3v, h2, nullptr, g_max, 1024);
  fc_kernel<true><<<16, 256, 0, stream>>>(
      (const float*)g_max, f1w, f1b, sb4g, sb4b, sb4m, sb4v, f1, 512, 1024, 0);
  fc_kernel<true><<<8, 256, 0, stream>>>(
      f1, f2w, f2b, sb5g, sb5b, sb5m, sb5v, f2, 256, 512, 0);
  fc_kernel<false><<<1, 128, 0, stream>>>(
      f2, f3w, f3b, nullptr, nullptr, nullptr, nullptr, out_trans, 9, 256, 1);

  // ---- main path (transform fused into conv1) ----
  conv1_bn_relu_kernel<true><<<256, 256, 0, stream>>>(
      x, out_trans, c1w, c1b, bn1g, bn1b, bn1m, bn1v, h1);
  wmma_pconv_kernel<128, 64, 0><<<dim3(512, 1), 256, 0, stream>>>(
      wbf_c2, c2b, bn2g, bn2b, bn2m, bn2v, h1, h2, nullptr, 128);
  wmma_pconv_kernel<64, 128, 2><<<dim3(512, 1), 256, 0, stream>>>(
      wbf_c3, c3b, bn3g, bn3b, bn3m, bn3v, h2, feats_t, nullptr, 64);

  // ---- SoftPool ----
  softpool_sort_kernel<<<512, 1024, 0, stream>>>(feats_t, idxbuf, out_spidx);
  softpool_gather_kernel<<<4096, 256, 0, stream>>>(feats_t, idxbuf, out_main);
}
